// Sae_21620865368722
// MI455X (gfx1250) — compile-verified, matrix operand order checked by reading, not run on hardware
//
#include <hip/hip_runtime.h>

#define N_ROWS 2048
#define D_IN   1024
#define L_FEAT 32768
#define K_SEL  32
#define MK     128

// ---- GEMM tiling ----
#define BM 64
#define BN 256
#define BK 32
#define NKSTEPS (D_IN / BK)
#define LDA 40   // padded LDS stride in bf16 elements (80B rows -> conflict-friendly)

typedef float v8f  __attribute__((ext_vector_type(8)));
typedef __bf16 v16bf __attribute__((ext_vector_type(16)));
typedef __bf16 v8bf  __attribute__((ext_vector_type(8)));
typedef unsigned short ushort4v __attribute__((ext_vector_type(4)));

union BF16Frag { v16bf v; v8bf h[2]; };

__device__ __forceinline__ unsigned short f2bf(float f) {
    unsigned int u = __float_as_uint(f);
    u += 0x7FFFu + ((u >> 16) & 1u);   // round-to-nearest-even
    return (unsigned short)(u >> 16);
}

// =====================================================================
// Kernel 1: pre_acts = relu((x - b_dec) @ enc_W^T + enc_b)   [N, L]
// bf16 WMMA, f32 accumulate.  Block tile 64x256, 8 waves of 32x64.
// Double-buffered LDS, one barrier per K-step, global loads for tile
// k+1 overlapped with WMMAs on tile k.
// =====================================================================
__global__ __launch_bounds__(256)
void sae_encoder_gemm(const float* __restrict__ x,
                      const float* __restrict__ encW,
                      const float* __restrict__ encB,
                      const float* __restrict__ bdec,
                      float* __restrict__ pre) {
    __shared__ unsigned short As[2][BM * LDA];
    __shared__ unsigned short Bs[2][BN * LDA];

    const int tid  = threadIdx.x;
    const int lane = tid & 31;
    const int wave = tid >> 5;
    const int wm   = wave >> 2;        // 0..1 (rows)
    const int wn   = wave & 3;         // 0..3 (cols)
    const int half = lane >> 4;        // K half per WMMA layout
    const int r    = lane & 15;

    const int m0 = blockIdx.y * BM;
    const int n0 = blockIdx.x * BN;

    // fill mappings
    const int arow = tid >> 2, aq = tid & 3;   // A: 64 rows x 2 float4 each
    const int brow = tid >> 3, bq = tid & 7;   // B: 8 iters of 32 rows x 8 float4

    const float* xbase = x    + (size_t)(m0 + arow) * D_IN + aq * 4;
    const float* bbase = bdec + aq * 4;
    const float* wbase = encW + (size_t)(n0 + brow) * D_IN + bq * 4;

    v8f acc[2][4];
#pragma unroll
    for (int i = 0; i < 2; ++i)
#pragma unroll
        for (int j = 0; j < 4; ++j)
#pragma unroll
            for (int v = 0; v < 8; ++v) acc[i][j][v] = 0.0f;

    // staging registers
    float4 ax0, ax1, bd0, bd1, bw[8];

    // ---- prologue: load + stage tile 0 ----
    ax0 = *(const float4*)(xbase);
    ax1 = *(const float4*)(xbase + 16);
    bd0 = *(const float4*)(bbase);
    bd1 = *(const float4*)(bbase + 16);
#pragma unroll
    for (int i = 0; i < 8; ++i)
        bw[i] = *(const float4*)(wbase + (size_t)(i * 32) * D_IN);
    {
        ushort4v s0 = { f2bf(ax0.x - bd0.x), f2bf(ax0.y - bd0.y),
                        f2bf(ax0.z - bd0.z), f2bf(ax0.w - bd0.w) };
        ushort4v s1 = { f2bf(ax1.x - bd1.x), f2bf(ax1.y - bd1.y),
                        f2bf(ax1.z - bd1.z), f2bf(ax1.w - bd1.w) };
        *(ushort4v*)&As[0][arow * LDA + aq * 4]      = s0;
        *(ushort4v*)&As[0][arow * LDA + 16 + aq * 4] = s1;
#pragma unroll
        for (int i = 0; i < 8; ++i) {
            ushort4v s = { f2bf(bw[i].x), f2bf(bw[i].y), f2bf(bw[i].z), f2bf(bw[i].w) };
            *(ushort4v*)&Bs[0][(brow + i * 32) * LDA + bq * 4] = s;
        }
    }
    __syncthreads();

    for (int kk = 0; kk < NKSTEPS; ++kk) {
        const int buf = kk & 1;
        const int k1  = (kk + 1) * BK;

        // ---- issue global loads for tile kk+1 (overlap with WMMA) ----
        if (kk + 1 < NKSTEPS) {
            ax0 = *(const float4*)(xbase + k1);
            ax1 = *(const float4*)(xbase + k1 + 16);
            bd0 = *(const float4*)(bbase + k1);
            bd1 = *(const float4*)(bbase + k1 + 16);
#pragma unroll
            for (int i = 0; i < 8; ++i)
                bw[i] = *(const float4*)(wbase + (size_t)(i * 32) * D_IN + k1);
            if (kk + 2 < NKSTEPS) {  // deep prefetch (global_prefetch_b8)
                __builtin_prefetch(xbase + k1 + BK, 0, 1);
                __builtin_prefetch(wbase + k1 + BK, 0, 1);
            }
        }

        // ---- fragments per ISA layout: lanes 0-15 K=[h*8..+7], [16+h*8..+7] ----
        BF16Frag a[2], b[4];
#pragma unroll
        for (int i = 0; i < 2; ++i) {
            const int row = wm * 32 + i * 16 + r;
            a[i].h[0] = *(const v8bf*)&As[buf][row * LDA + half * 8];
            a[i].h[1] = *(const v8bf*)&As[buf][row * LDA + 16 + half * 8];
        }
#pragma unroll
        for (int j = 0; j < 4; ++j) {
            const int col = wn * 64 + j * 16 + r;
            b[j].h[0] = *(const v8bf*)&Bs[buf][col * LDA + half * 8];
            b[j].h[1] = *(const v8bf*)&Bs[buf][col * LDA + 16 + half * 8];
        }
#pragma unroll
        for (int i = 0; i < 2; ++i)
#pragma unroll
            for (int j = 0; j < 4; ++j)
                acc[i][j] = __builtin_amdgcn_wmma_f32_16x16x32_bf16(
                    false, a[i].v, false, b[j].v, (short)0, acc[i][j], false, false);

        // ---- stage tile kk+1 into the other buffer; single barrier ----
        if (kk + 1 < NKSTEPS) {
            const int nb = buf ^ 1;
            ushort4v s0 = { f2bf(ax0.x - bd0.x), f2bf(ax0.y - bd0.y),
                            f2bf(ax0.z - bd0.z), f2bf(ax0.w - bd0.w) };
            ushort4v s1 = { f2bf(ax1.x - bd1.x), f2bf(ax1.y - bd1.y),
                            f2bf(ax1.z - bd1.z), f2bf(ax1.w - bd1.w) };
            *(ushort4v*)&As[nb][arow * LDA + aq * 4]      = s0;
            *(ushort4v*)&As[nb][arow * LDA + 16 + aq * 4] = s1;
#pragma unroll
            for (int i = 0; i < 8; ++i) {
                ushort4v s = { f2bf(bw[i].x), f2bf(bw[i].y), f2bf(bw[i].z), f2bf(bw[i].w) };
                *(ushort4v*)&Bs[nb][(brow + i * 32) * LDA + bq * 4] = s;
            }
            __syncthreads();
        }
    }

    // ---- epilogue: + enc_b, relu, store f32 ----
#pragma unroll
    for (int j = 0; j < 4; ++j) {
        const int gn = n0 + wn * 64 + j * 16 + r;
        const float bj = encB[gn];
#pragma unroll
        for (int i = 0; i < 2; ++i) {
#pragma unroll
            for (int v = 0; v < 8; ++v) {
                const int gm = m0 + wm * 32 + i * 16 + half * 8 + v;
                float val = acc[i][j][v] + bj;
                pre[(size_t)gm * L_FEAT + gn] = val > 0.0f ? val : 0.0f;
            }
        }
    }
}

// =====================================================================
// Kernel 2: per-row top-128 (sorted desc) via LDS-resident iterative max
// =====================================================================
__global__ __launch_bounds__(256)
void sae_topk(const float* __restrict__ pre,
              float* __restrict__ out_acts, float* __restrict__ out_idx,
              float* __restrict__ ws_acts, int* __restrict__ ws_idx) {
    extern __shared__ float smem[];
    float* vals = smem;                 // L_FEAT floats (128 KB)
    float* rv   = smem + L_FEAT;        // 256
    int*   ri   = (int*)(rv + 256);     // 256

    const int n   = blockIdx.x;
    const int tid = threadIdx.x;
    const float* row = pre + (size_t)n * L_FEAT;
    for (int i = tid; i < L_FEAT; i += 256) vals[i] = row[i];
    __syncthreads();

    for (int k = 0; k < MK; ++k) {
        float bv = -3.4e38f; int bi = 0x7FFFFFFF;
        for (int i = tid; i < L_FEAT; i += 256) {
            float v = vals[i];
            if (v > bv) { bv = v; bi = i; }
        }
        rv[tid] = bv; ri[tid] = bi;
        __syncthreads();
        for (int s = 128; s > 0; s >>= 1) {
            if (tid < s) {
                float ov = rv[tid + s]; int oi = ri[tid + s];
                if (ov > rv[tid] || (ov == rv[tid] && oi < ri[tid])) {
                    rv[tid] = ov; ri[tid] = oi;
                }
            }
            __syncthreads();
        }
        if (tid == 0) {
            float v = rv[0]; int i = ri[0];
            out_acts[(size_t)n * MK + k] = v;
            out_idx [(size_t)n * MK + k] = __int_as_float(i);  // int32 bit pattern
            ws_acts [(size_t)n * MK + k] = v;
            ws_idx  [(size_t)n * MK + k] = i;
            vals[i] = -3.4e38f;
        }
        __syncthreads();
    }
}

// =====================================================================
// Kernel 3: column sums + total sum of squares of x (for total_variance)
// =====================================================================
__global__ __launch_bounds__(256)
void sae_colstats(const float* __restrict__ x,
                  float* __restrict__ colsum, float* __restrict__ sumsq) {
    __shared__ float s1[256], s2[256];
    const int col = blockIdx.x, tid = threadIdx.x;
    float a = 0.0f, b = 0.0f;
    for (int n = tid; n < N_ROWS; n += 256) {
        float v = x[(size_t)n * D_IN + col];
        a += v; b += v * v;
    }
    s1[tid] = a; s2[tid] = b;
    __syncthreads();
    for (int s = 128; s > 0; s >>= 1) {
        if (tid < s) { s1[tid] += s1[tid + s]; s2[tid] += s2[tid + s]; }
        __syncthreads();
    }
    if (tid == 0) { colsum[col] = s1[0]; atomicAdd(sumsq, s2[0]); }
}

// =====================================================================
// Kernel 4: sparse decode (k=32 and k=128 fused) + error norms
// =====================================================================
__global__ __launch_bounds__(256)
void sae_decode(const float* __restrict__ x, const float* __restrict__ Wdec,
                const float* __restrict__ bdec,
                const float* __restrict__ ws_acts, const int* __restrict__ ws_idx,
                float* __restrict__ sae_out,
                float* __restrict__ num32, float* __restrict__ num128) {
    __shared__ float sa[MK];
    __shared__ int   si[MK];
    __shared__ float red[256];
    const int n = blockIdx.x, tid = threadIdx.x;
    if (tid < MK) {
        sa[tid] = ws_acts[(size_t)n * MK + tid];
        si[tid] = ws_idx [(size_t)n * MK + tid];
    }
    __syncthreads();

    float4 y32 = {0, 0, 0, 0}, y128 = {0, 0, 0, 0};
#pragma unroll 4
    for (int k = 0; k < K_SEL; ++k) {
        const float a = sa[k];
        const float4 w = *(const float4*)(Wdec + (size_t)si[k] * D_IN + tid * 4);
        y32.x += a * w.x; y32.y += a * w.y; y32.z += a * w.z; y32.w += a * w.w;
        y128.x += a * w.x; y128.y += a * w.y; y128.z += a * w.z; y128.w += a * w.w;
    }
#pragma unroll 4
    for (int k = K_SEL; k < MK; ++k) {
        const float a = sa[k];
        const float4 w = *(const float4*)(Wdec + (size_t)si[k] * D_IN + tid * 4);
        y128.x += a * w.x; y128.y += a * w.y; y128.z += a * w.z; y128.w += a * w.w;
    }
    const float4 bd = *(const float4*)(bdec + tid * 4);
    const float4 xv = *(const float4*)(x + (size_t)n * D_IN + tid * 4);
    y32.x += bd.x; y32.y += bd.y; y32.z += bd.z; y32.w += bd.w;
    y128.x += bd.x; y128.y += bd.y; y128.z += bd.z; y128.w += bd.w;
    *(float4*)(sae_out + (size_t)n * D_IN + tid * 4) = y128;

    float e32 = 0.0f, e128 = 0.0f;
    { float d; d = y32.x - xv.x; e32 += d * d; d = y32.y - xv.y; e32 += d * d;
      d = y32.z - xv.z; e32 += d * d; d = y32.w - xv.w; e32 += d * d;
      d = y128.x - xv.x; e128 += d * d; d = y128.y - xv.y; e128 += d * d;
      d = y128.z - xv.z; e128 += d * d; d = y128.w - xv.w; e128 += d * d; }

    red[tid] = e32; __syncthreads();
    for (int s = 128; s > 0; s >>= 1) { if (tid < s) red[tid] += red[tid + s]; __syncthreads(); }
    if (tid == 0) atomicAdd(num32, red[0]);
    __syncthreads();
    red[tid] = e128; __syncthreads();
    for (int s = 128; s > 0; s >>= 1) { if (tid < s) red[tid] += red[tid + s]; __syncthreads(); }
    if (tid == 0) atomicAdd(num128, red[0]);
}

// =====================================================================
// Kernel 5: scalars    Kernel 0: zero accumulators
// =====================================================================
__global__ void sae_init(float* __restrict__ scal) {
    if (threadIdx.x < 4) scal[threadIdx.x] = 0.0f;
}

__global__ __launch_bounds__(256)
void sae_finalize(const float* __restrict__ colsum, const float* __restrict__ scal,
                  float* __restrict__ out_scalars) {
    __shared__ float red[256];
    const int tid = threadIdx.x;
    float s = 0.0f;
    for (int d = tid; d < D_IN; d += 256) { float c = colsum[d]; s += c * c; }
    red[tid] = s; __syncthreads();
    for (int st = 128; st > 0; st >>= 1) { if (tid < st) red[tid] += red[tid + st]; __syncthreads(); }
    if (tid == 0) {
        const float tv = scal[0] - red[0] / (float)N_ROWS; // sum x^2 - sum_d colsum^2/N
        out_scalars[0] = scal[1] / tv;   // fvu (k=32)
        out_scalars[1] = 0.0f;           // auxk_loss
        out_scalars[2] = scal[2] / tv;   // multi_topk_fvu (k=128)
    }
}

extern "C" void kernel_launch(void* const* d_in, const int* in_sizes, int n_in,
                              void* d_out, int out_size, void* d_ws, size_t ws_size,
                              hipStream_t stream) {
    const float* x    = (const float*)d_in[0];
    const float* encW = (const float*)d_in[1];
    const float* encB = (const float*)d_in[2];
    const float* Wdec = (const float*)d_in[3];
    const float* bdec = (const float*)d_in[4];

    // workspace layout
    float* ws_pre    = (float*)d_ws;                                   // N*L
    float* ws_acts   = ws_pre + (size_t)N_ROWS * L_FEAT;               // N*MK
    int*   ws_idx    = (int*)(ws_acts + (size_t)N_ROWS * MK);          // N*MK
    float* ws_colsum = (float*)(ws_idx + (size_t)N_ROWS * MK);         // D_IN
    float* ws_scal   = ws_colsum + D_IN;                               // [sumsq, num32, num128, pad]

    // d_out layout: sae_out | top_acts | top_indices | fvu | auxk | multi_fvu
    float* out_sae     = (float*)d_out;
    float* out_acts    = out_sae + (size_t)N_ROWS * D_IN;
    float* out_idx     = out_acts + (size_t)N_ROWS * MK;
    float* out_scalars = out_idx + (size_t)N_ROWS * MK;

    sae_init<<<1, 32, 0, stream>>>(ws_scal);

    dim3 g1(L_FEAT / BN, N_ROWS / BM);   // (128, 32)
    sae_encoder_gemm<<<g1, 256, 0, stream>>>(x, encW, encB, bdec, ws_pre);

    size_t shmem = (size_t)(L_FEAT + 256) * sizeof(float) + 256 * sizeof(int);
    sae_topk<<<N_ROWS, 256, shmem, stream>>>(ws_pre, out_acts, out_idx, ws_acts, ws_idx);

    sae_colstats<<<D_IN, 256, 0, stream>>>(x, ws_colsum, &ws_scal[0]);

    sae_decode<<<N_ROWS, 256, 0, stream>>>(x, Wdec, bdec, ws_acts, ws_idx,
                                           out_sae, &ws_scal[1], &ws_scal[2]);

    sae_finalize<<<1, 256, 0, stream>>>(ws_colsum, ws_scal, out_scalars);
}
